// AttentionLayer_3324304687704
// MI455X (gfx1250) — matrix operand order repaired and checked by reading, not compiled, not run on hardware
//
#include <hip/hip_runtime.h>
#include <hip/hip_bf16.h>

// Problem constants (from reference): B=16, S=4096, D=1024, H=16, hd=64
#define BB 16
#define SS 4096
#define DD 1024
#define HH 16
#define HD 64

typedef __attribute__((ext_vector_type(16))) __bf16 v16bf;
typedef __attribute__((ext_vector_type(8)))  __bf16 v8bf;
typedef __attribute__((ext_vector_type(8)))  float  v8f;
typedef __attribute__((ext_vector_type(4)))  int    v4i;

__device__ __forceinline__ v16bf cat8(v8bf lo, v8bf hi) {
  return __builtin_shufflevector(lo, hi, 0,1,2,3,4,5,6,7,8,9,10,11,12,13,14,15);
}

// Convert 16 contiguous f32 in LDS -> bf16 fragment (compiler emits ds_load_b128
// + v_cvt_pk_bf16_f32 pairs).
__device__ __forceinline__ v16bf load16f_bf(const float* p) {
  v16bf r;
  #pragma unroll
  for (int i = 0; i < 16; ++i) r[i] = (__bf16)p[i];
  return r;
}

// ---------------------------------------------------------------------------
// Async global -> LDS staging (CDNA5 GLOBAL_LOAD_ASYNC_TO_LDS_*, ASYNCcnt).
// Builtin signatures (from hipcc diagnostics): b128 = (AS1 v4i*, AS3 v4i*,
// imm offset, imm cpol); b32 = (AS1 int*, AS3 int*, imm, imm).
// Guarded: falls back to synchronous copies if the builtins are absent.
// ---------------------------------------------------------------------------
#if defined(__gfx1250__) && __has_builtin(__builtin_amdgcn_global_load_async_to_lds_b128)
#define HAVE_ASYNC_LDS 1
#else
#define HAVE_ASYNC_LDS 0
#endif

__device__ __forceinline__ void stage_b128(const float* g, float* l) {
#if HAVE_ASYNC_LDS
  __builtin_amdgcn_global_load_async_to_lds_b128(
      (__attribute__((address_space(1))) v4i*)(uintptr_t)g,
      (__attribute__((address_space(3))) v4i*)(uint32_t)(uintptr_t)l, 0, 0);
#else
  *(float4*)l = *(const float4*)g;
#endif
}

__device__ __forceinline__ void stage_b32(const float* g, float* l) {
#if HAVE_ASYNC_LDS
  __builtin_amdgcn_global_load_async_to_lds_b32(
      (__attribute__((address_space(1))) int*)(uintptr_t)g,
      (__attribute__((address_space(3))) int*)(uint32_t)(uintptr_t)l, 0, 0);
#else
  *l = *g;
#endif
}

__device__ __forceinline__ void stage_wait() {
#if HAVE_ASYNC_LDS
#if __has_builtin(__builtin_amdgcn_s_wait_asynccnt)
  __builtin_amdgcn_s_wait_asynccnt(0);
#else
  asm volatile("s_wait_asynccnt 0x0" ::: "memory");
#endif
#endif
}

// ---------------------------------------------------------------------------
// Kernel 1: q[b,i] = dec[b,:] . Wq[i,:]        (16x1024 outputs, 33 MFLOP)
// ---------------------------------------------------------------------------
__global__ void k_qproj(const float* __restrict__ dec, const float* __restrict__ Wq,
                        float* __restrict__ q) {
  int idx = blockIdx.x * blockDim.x + threadIdx.x;   // 16384
  int b = idx >> 10, i = idx & (DD - 1);
  const float4* x = (const float4*)(dec + (size_t)b * DD);
  const float4* w = (const float4*)(Wq + (size_t)i * DD);
  float acc = 0.f;
  #pragma unroll 4
  for (int d4 = 0; d4 < DD / 4; ++d4) {
    float4 a = x[d4], c = w[d4];
    acc += a.x * c.x + a.y * c.y + a.z * c.z + a.w * c.w;
  }
  q[idx] = acc;
}

// ---------------------------------------------------------------------------
// Kernel 2: qk[b,h,d] = scale * sum_j q[b,h*64+j] * Wk[h*64+j, d]  -> bf16
// Folds Wk into the query (query length == 1), scale = 1/sqrt(64) = 0.125
// ---------------------------------------------------------------------------
__global__ void k_qk(const float* __restrict__ q, const float* __restrict__ Wk,
                     __bf16* __restrict__ qkbf) {
  int idx = blockIdx.x * blockDim.x + threadIdx.x;  // B*H*D = 262144
  int d = idx & (DD - 1);
  int h = (idx >> 10) & (HH - 1);
  int b = idx >> 14;
  const float* qrow = q + (size_t)b * DD + h * HD;
  float acc = 0.f;
  __builtin_prefetch(&Wk[(size_t)(h * HD) * DD + d], 0, 0);
  #pragma unroll 8
  for (int j = 0; j < HD; ++j)
    acc += qrow[j] * Wk[(size_t)(h * HD + j) * DD + d];
  qkbf[idx] = (__bf16)(acc * 0.125f);
}

// ---------------------------------------------------------------------------
// Kernel 3: scores[b,h,s] = qk[b,h,:] . X[b,s,:]   via v_wmma_f32_16x16x32_bf16
// Grid: (S/64, B).  128 threads = 4 waves; wave w owns s-tile [w*16, w*16+16).
// A = qk (M=16 heads x K), B = X^T (K x N=16 s)  -> B column n == X row s_n,
// so each lane's B fragment is a contiguous K-run: no transpose needed.
// Double-buffered async staging: stage tile kc+1 into LDS while computing kc.
// ---------------------------------------------------------------------------
#define XF3 68   // f32 LDS row stride: 272B = 17*16B, conflict-free across 16 lanes
__device__ __forceinline__ void stage_tile_scores(const float* xg, int kc,
                                                  float (*dst)[XF3], int t) {
  #pragma unroll
  for (int i = 0; i < 8; ++i) {
    int f = t + i * 128;               // 1024 x 16B = 64x64 f32
    int row = f >> 4, d4 = (f & 15) * 4;
    stage_b128(xg + (size_t)row * DD + kc * 64 + d4, &dst[row][d4]);
  }
}

__global__ void __launch_bounds__(128)
k_scores(const float* __restrict__ X, const __bf16* __restrict__ qkbf,
         float* __restrict__ scores) {
  __shared__ __align__(16) float xf[2][64][XF3];    // 2 x 17 KB
  const int t  = threadIdx.x;
  const int b  = blockIdx.y;
  const int s0 = blockIdx.x * 64;
  const int lane = t & 31, w = t >> 5;
  const int nl = lane & 15, hi = lane >> 4;
  const float*  xg = X + ((size_t)b * SS + s0) * DD;
  const __bf16* ap = qkbf + (size_t)(b * HH + nl) * DD;  // A row = head nl (L2-hot)
  v8f acc = {};

  stage_tile_scores(xg, 0, xf[0], t);
  stage_wait();
  __syncthreads();

  int buf = 0;
  for (int kc = 0; kc < DD / 64; ++kc) {
    if (kc + 1 < DD / 64) stage_tile_scores(xg, kc + 1, xf[buf ^ 1], t);

    #pragma unroll
    for (int sub = 0; sub < 2; ++sub) {      // two K=32 steps per 64-d chunk
      int kbase = kc * 64 + sub * 32;
      // A fragment (contiguous K order, matching B): lane (nl,hi) holds
      // row h=nl, K = {hi*16..+15}
      v8bf alo = *(const v8bf*)(ap + kbase + hi * 16);
      v8bf ahi = *(const v8bf*)(ap + kbase + hi * 16 + 8);
      v16bf a = cat8(alo, ahi);
      // B fragment: lane n holds column n (= X row s0+w*16+n), K-run of 16
      v16bf bv = load16f_bf(&xf[buf][w * 16 + nl][sub * 32 + hi * 16]);
      acc = __builtin_amdgcn_wmma_f32_16x16x32_bf16(false, a, false, bv,
                                                    (short)0, acc, false, false);
    }

    if (kc + 1 < DD / 64) stage_wait();
    __syncthreads();
    buf ^= 1;
  }

  // C/D layout: VGPR r -> M = hi*8 + r, N = nl
  #pragma unroll
  for (int r = 0; r < 8; ++r)
    scores[((size_t)(b * HH) + hi * 8 + r) * SS + s0 + w * 16 + nl] = acc[r];
}

// ---------------------------------------------------------------------------
// Kernel 4: softmax over S per (b,h); writes normalized weights as bf16.
// ---------------------------------------------------------------------------
__global__ void __launch_bounds__(256)
k_softmax(const float* __restrict__ scores, __bf16* __restrict__ pbf) {
  __shared__ float red[256];
  const int bh = blockIdx.x;            // 0..255
  const int t  = threadIdx.x;
  const float* sc = scores + (size_t)bh * SS;
  float m = -3.4e38f;
  for (int i = t; i < SS; i += 256) m = fmaxf(m, sc[i]);
  red[t] = m; __syncthreads();
  for (int o = 128; o > 0; o >>= 1) { if (t < o) red[t] = fmaxf(red[t], red[t + o]); __syncthreads(); }
  m = red[0]; __syncthreads();
  float l = 0.f;
  for (int i = t; i < SS; i += 256) l += __expf(sc[i] - m);
  red[t] = l; __syncthreads();
  for (int o = 128; o > 0; o >>= 1) { if (t < o) red[t] += red[t + o]; __syncthreads(); }
  l = red[0];
  float inv = 1.0f / l;
  for (int i = t; i < SS; i += 256)
    pbf[(size_t)bh * SS + i] = (__bf16)(__expf(sc[i] - m) * inv);
}

// ---------------------------------------------------------------------------
// Kernel 5: u[b,h,d] = sum_s p[b,h,s] * X[b,s,d]  via WMMA.
// Grid: (D/64, B). 4 waves; wave w owns d-tile [w*16, w*16+16).
// A = p (M=16 heads x K=s), B = X (K=s x N=d). B column n = X[:,d_n] is
// strided in memory, so the async engine scatters the 32x64 chunk transposed
// into LDS element-wise (global side stays coalesced: lanes walk d).
// ---------------------------------------------------------------------------
#define XU 36   // f32 LDS row stride: 144B = 9*16B, conflict-free across 16 lanes
__device__ __forceinline__ void stage_tile_u(const float* xg, int scnk,
                                             float (*dst)[XU], int t) {
  #pragma unroll
  for (int i = 0; i < 16; ++i) {
    int e = t + i * 128;               // 2048 f32 = 32s x 64d
    int d = e & 63, s = e >> 6;
    stage_b32(xg + (size_t)(scnk * 32 + s) * DD + d, &dst[d][s]);
  }
}

__global__ void __launch_bounds__(128)
k_u(const float* __restrict__ X, const __bf16* __restrict__ pbf,
    float* __restrict__ u) {
  __shared__ __align__(16) float xu[2][64][XU];     // 2 x 9 KB, xu[..][d][s]
  const int t = threadIdx.x;
  const int b = blockIdx.y;
  const int dbase = blockIdx.x * 64;
  const int lane = t & 31, w = t >> 5;
  const int nl = lane & 15, hi = lane >> 4;
  const float*  xg   = X + (size_t)b * SS * DD + dbase;
  const __bf16* prow = pbf + (size_t)(b * HH + nl) * SS;  // A row = head nl
  v8f acc = {};

  stage_tile_u(xg, 0, xu[0], t);
  stage_wait();
  __syncthreads();

  int buf = 0;
  for (int scnk = 0; scnk < SS / 32; ++scnk) {
    if (scnk + 1 < SS / 32) stage_tile_u(xg, scnk + 1, xu[buf ^ 1], t);

    {
      int kb = scnk * 32;
      v8bf alo = *(const v8bf*)(prow + kb + hi * 16);
      v8bf ahi = *(const v8bf*)(prow + kb + hi * 16 + 8);
      v16bf a = cat8(alo, ahi);
      v16bf bv = load16f_bf(&xu[buf][w * 16 + nl][hi * 16]);
      acc = __builtin_amdgcn_wmma_f32_16x16x32_bf16(false, a, false, bv,
                                                    (short)0, acc, false, false);
    }

    if (scnk + 1 < SS / 32) stage_wait();
    __syncthreads();
    buf ^= 1;
  }

  #pragma unroll
  for (int r = 0; r < 8; ++r)
    u[(size_t)(b * HH + hi * 8 + r) * DD + dbase + w * 16 + nl] = acc[r];
}

// ---------------------------------------------------------------------------
// Kernel 6: out[b, h*64+j] = Wv[h*64+j, :] . u[b,h,:]
// ---------------------------------------------------------------------------
__global__ void k_ctx(const float* __restrict__ Wv, const float* __restrict__ u,
                      float* __restrict__ out) {
  int idx = blockIdx.x * blockDim.x + threadIdx.x;   // 16384
  int b = idx >> 10, i = idx & (DD - 1);
  int h = i >> 6;
  const float4* wv = (const float4*)(Wv + (size_t)i * DD);
  const float4* ur = (const float4*)(u + (size_t)(b * HH + h) * DD);
  float acc = 0.f;
  #pragma unroll 4
  for (int d4 = 0; d4 < DD / 4; ++d4) {
    float4 a = wv[d4], c = ur[d4];
    acc += a.x * c.x + a.y * c.y + a.z * c.z + a.w * c.w;
  }
  out[idx] = acc;
}

// ---------------------------------------------------------------------------
extern "C" void kernel_launch(void* const* d_in, const int* in_sizes, int n_in,
                              void* d_out, int out_size, void* d_ws, size_t ws_size,
                              hipStream_t stream) {
  const float* dec = (const float*)d_in[0];   // [16,1024]
  const float* X   = (const float*)d_in[1];   // [16,4096,1024]
  const float* Wq  = (const float*)d_in[2];   // [1024,1024]
  const float* Wk  = (const float*)d_in[3];
  const float* Wv  = (const float*)d_in[4];
  float* out = (float*)d_out;                 // [16,1024]

  // Workspace layout (~7.9 MB total)
  float*  q      = (float*)d_ws;                       // 16384 f32
  float*  scores = q + BB * DD;                        // 1,048,576 f32
  float*  u      = scores + (size_t)BB * HH * SS;      // 262,144 f32
  __bf16* qkbf   = (__bf16*)(u + BB * HH * DD);        // 262,144 bf16
  __bf16* pbf    = qkbf + (size_t)BB * HH * DD;        // 1,048,576 bf16

  k_qproj  <<<BB * DD / 256, 256, 0, stream>>>(dec, Wq, q);
  k_qk     <<<BB * HH * DD / 256, 256, 0, stream>>>(q, Wk, qkbf);
  k_scores <<<dim3(SS / 64, BB), 128, 0, stream>>>(X, qkbf, scores);
  k_softmax<<<BB * HH, 256, 0, stream>>>(scores, pbf);
  k_u      <<<dim3(DD / 64, BB), 128, 0, stream>>>(X, pbf, u);
  k_ctx    <<<BB * DD / 256, 256, 0, stream>>>(Wv, u, out);
}
